// SurfaceContinuousKernelPositionalEmbedding_9156870275550
// MI455X (gfx1250) — matrix operand order repaired and checked by reading, not compiled, not run on hardware
//
#include <hip/hip_runtime.h>
#include <hip/hip_bf16.h>

typedef __attribute__((ext_vector_type(16))) _Float16 v16h;
typedef __attribute__((ext_vector_type(8)))  float    v8f;

#define LOG2E  1.4426950408889634f
#define SQRT2F 1.41421356237309515f

// One workgroup = one (batch b, 16-row tile of the N=512 concatenated rows).
// 4 waves per block; wave w owns d in [16w, 16w+16).
__global__ __launch_bounds__(128)
void nw_kernel_embed(const float* __restrict__ surf_lm,
                     const float* __restrict__ surf_ttm,
                     const float* __restrict__ surf_tv,
                     const float* __restrict__ q_lm,
                     const float* __restrict__ q_ttm,
                     const float* __restrict__ kls,
                     const float* __restrict__ ln_s_w,
                     const float* __restrict__ ln_s_b,
                     const float* __restrict__ ln_q_w,
                     const float* __restrict__ ln_q_b,
                     const float* __restrict__ pe_log_scale,
                     float* __restrict__ out)
{
    constexpr int S = 256, D = 64, BATCH = 8;

    __shared__ float2   dsq[S * 16];      // (dx^2, dt^2) indexed [s*16 + n]  (32 KB)
    __shared__ float    sx[S], st[S];     // surface coords
    __shared__ _Float16 tvh[S];           // surf_tv as f16 (B-matrix column 0)
    __shared__ float    cx[16], ct[16];   // tile-row coords
    __shared__ float    a2[D], b2[D];     // folded per-d constants (base-2 exponent coeffs)
    __shared__ float    wsum_s[16][D];
    __shared__ float    norm_s[16][D];
    __shared__ float    part[16][8][2];   // per-row partial (sum, sumsq) for layernorm

    const int t    = threadIdx.x;
    const int b    = blockIdx.x >> 5;     // 32 tiles per batch
    const int tile = blockIdx.x & 31;
    const int n0   = tile * 16;           // global row base within N=512

    // ---- Phase A: stage coords / tv / per-d coefficients --------------------
    for (int i = t; i < S; i += 128) {
        sx[i]  = surf_lm[b * S + i];
        st[i]  = surf_ttm[b * S + i];
        tvh[i] = (_Float16)surf_tv[b * S + i];
    }
    if (t < 16) {
        int n = n0 + t;
        if (n < S) { cx[t] = surf_lm[b * S + n];     ct[t] = surf_ttm[b * S + n]; }
        else       { cx[t] = q_lm[b * S + (n - S)];  ct[t] = q_ttm[b * S + (n - S)]; }
    }
    if (t < D) {
        float s0 = __expf(kls[2 * t]);
        float s1 = __expf(kls[2 * t + 1]);
        float bw = erfinvf((float)(t + 1) * (1.0f / 65.0f)) * SQRT2F;
        // kv = exp(-(dx2*s0+dt2*s1)/(tr*2bw^2)) = exp2(dx2*a2 + dt2*b2)
        float inv = -LOG2E / ((s0 + s1) * 2.0f * bw * bw);
        a2[t] = s0 * inv;
        b2[t] = s1 * inv;
    }
    __syncthreads();

    // ---- Phase B: (dx^2, dt^2) for all (s, n) pairs, reused across all 64 d -
    for (int i = t; i < S * 16; i += 128) {
        int s = i >> 4, n = i & 15;
        float dx = cx[n] - sx[s];
        float dt = ct[n] - st[s];
        dsq[i] = make_float2(dx * dx, dt * dt);
    }
    __syncthreads();

    // ---- Phase C: per-wave WMMA reduction over s ----------------------------
    const int wave = t >> 5;
    const int lane = t & 31;
    const int m    = lane & 15;   // A row M / B,C column N
    const int h    = lane >> 4;   // which half of the wave

    // B matrices per s-chunk: [32 x 16], col 0 = tv, col 1 = 1, rest 0.
    // k order used here (contiguous, same as A): element e -> K = h*16 + e, column N = lane & 15.
    v16h Bm[8];
#pragma unroll
    for (int c = 0; c < 8; ++c) {
        v16h bv;
#pragma unroll
        for (int e = 0; e < 16; ++e) {
            _Float16 v = (_Float16)0.0f;
            if      (m == 0) v = tvh[c * 32 + h * 16 + e];
            else if (m == 1) v = (_Float16)1.0f;
            bv[e] = v;
        }
        Bm[c] = bv;
    }

    for (int dd = 0; dd < 16; ++dd) {
        int   d  = wave * 16 + dd;
        float ad = a2[d], bd = b2[d];
        v8f acc = {0.f, 0.f, 0.f, 0.f, 0.f, 0.f, 0.f, 0.f};
#pragma unroll
        for (int c = 0; c < 8; ++c) {
            // A = kv[16 rows x 32 s].  same contiguous k order as B:
            // element e -> K = h*16 + e, row = lane&15.
            v16h Av;
#pragma unroll
            for (int e = 0; e < 16; ++e) {
                int k = h * 16 + e;
                float2 q = dsq[(c * 32 + k) * 16 + m];
                float  kv = __builtin_amdgcn_exp2f(fmaf(q.x, ad, q.y * bd));
                Av[e] = (_Float16)kv;
            }
            acc = __builtin_amdgcn_wmma_f32_16x16x32_f16(
                      false, Av, false, Bm[c], (short)0, acc, false, false);
        }
        // C layout: VGPR r, lanes 0-15 -> (M=r, N=lane); lanes 16-31 -> (M=8+r, N=lane-16)
        if (m == 0) {                 // column 0 = weighted sum
#pragma unroll
            for (int r = 0; r < 8; ++r) wsum_s[h * 8 + r][d] = acc[r];
        } else if (m == 1) {          // column 1 = normalizer
#pragma unroll
            for (int r = 0; r < 8; ++r) norm_s[h * 8 + r][d] = acc[r];
        }
    }
    __syncthreads();

    // ---- Phase D: emb = wsum/norm, layernorm over D, + sqrt(2)*PE -----------
    // All 128 threads: 16 rows x 8 threads/row, 8 d's per thread.
    {
        const int  r       = t >> 3;        // row within tile
        const int  g       = t & 7;         // d-group within row
        const int  n       = n0 + r;
        const bool is_surf = (n < S);

        float e8[8];
        float ps = 0.0f, pss = 0.0f;
#pragma unroll
        for (int i = 0; i < 8; ++i) {
            int   d = g * 8 + i;
            float e = wsum_s[r][d] / norm_s[r][d];
            e8[i] = e;
            ps  += e;
            pss += e * e;
        }
        part[r][g][0] = ps;
        part[r][g][1] = pss;
        __syncthreads();

        float sum = 0.0f, sumsq = 0.0f;
#pragma unroll
        for (int g2 = 0; g2 < 8; ++g2) {
            sum   += part[r][g2][0];
            sumsq += part[r][g2][1];
        }
        float mean = sum * (1.0f / D);
        float var  = sumsq * (1.0f / D) - mean * mean;   // biased variance
        float rstd = rsqrtf(fmaxf(var, 0.0f) + 1e-5f);

        const float* lw = is_surf ? ln_s_w : ln_q_w;
        const float* lb = is_surf ? ln_s_b : ln_q_b;
        float x   = cx[r], tc = ct[r];
        float pls = pe_log_scale[0];
        size_t off = is_surf ? ((size_t)(b * S + n) * D)
                             : ((size_t)BATCH * S * D + (size_t)(b * S + (n - S)) * D);
#pragma unroll
        for (int i = 0; i < 8; ++i) {
            int   d   = g * 8 + i;
            int   k4  = d >> 2, j = d & 3;
            float inv = __expf(-pls * (float)k4 * (1.0f / 16.0f));
            float arg = (j < 2 ? x : tc) * inv;
            float pe  = (j & 1) ? __cosf(arg) : __sinf(arg);
            out[off + d] = (e8[i] - mean) * rstd * lw[d] + lb[d] + pe * SQRT2F;
        }
    }
}

extern "C" void kernel_launch(void* const* d_in, const int* in_sizes, int n_in,
                              void* d_out, int out_size, void* d_ws, size_t ws_size,
                              hipStream_t stream) {
    (void)in_sizes; (void)n_in; (void)out_size; (void)d_ws; (void)ws_size;
    const float* surf_lm  = (const float*)d_in[0];
    const float* surf_ttm = (const float*)d_in[1];
    const float* surf_tv  = (const float*)d_in[2];
    const float* q_lm     = (const float*)d_in[3];
    const float* q_ttm    = (const float*)d_in[4];
    const float* kls      = (const float*)d_in[5];
    const float* ln_s_w   = (const float*)d_in[6];
    const float* ln_s_b   = (const float*)d_in[7];
    const float* ln_q_w   = (const float*)d_in[8];
    const float* ln_q_b   = (const float*)d_in[9];
    const float* pls      = (const float*)d_in[10];
    float* out = (float*)d_out;

    // grid: 8 batches * 32 tiles of 16 rows (N = 512); 128 threads = 4 waves
    nw_kernel_embed<<<dim3(256), dim3(128), 0, stream>>>(
        surf_lm, surf_ttm, surf_tv, q_lm, q_ttm, kls,
        ln_s_w, ln_s_b, ln_q_w, ln_q_b, pls, out);
}
